// TUPESelfAttention_90323162235484
// MI455X (gfx1250) — matrix operand order repaired and checked by reading, not compiled, not run on hardware
//
#include <hip/hip_runtime.h>
#include <hip/hip_bf16.h>

#define B_  4
#define T_  2048
#define D_  1024
#define H_  16
#define DK_ 64

typedef __attribute__((ext_vector_type(16))) __bf16 v16bf;
typedef __attribute__((ext_vector_type(8)))  float  v8f;

union Frag {
    v16bf v;
    uint4 u[2];
    unsigned short h[16];
};

__device__ __forceinline__ unsigned short f2bf(float f) {
    unsigned int u = __float_as_uint(f);
    u += 0x7FFFu + ((u >> 16) & 1u);     // round-to-nearest-even
    return (unsigned short)(u >> 16);
}

__device__ __forceinline__ v8f wmma_bf16(const Frag& a, const Frag& b, v8f c) {
    return __builtin_amdgcn_wmma_f32_16x16x32_bf16(false, a.v, false, b.v,
                                                   (short)0, c, false, false);
}

// ---------------------------------------------------------------- conversion
__global__ void conv_bf16_kernel(const float* __restrict__ src,
                                 unsigned short* __restrict__ dst, int n) {
    int i = blockIdx.x * blockDim.x + threadIdx.x;
    if (i < n) dst[i] = f2bf(src[i]);
}

// ---------------------------------------------------------------- GEMM
// C[m,n] = sum_k A[m,k] * W[n,k]   (A: MxK bf16 row-major, W: NxK bf16 row-major)
// Wave tile: 32(m) x 64(n).  Block: 8 waves = 64(m) x 256(n).
// mode 0: f32 out[m*N+n]
// mode 1: bf16 q layout  [B,H,T,DK], val*scale
// mode 2: bf16 k layout  [B,H,T,DK]
// mode 3: bf16 v layout  [B,H,DK,T] (transposed)
__global__ __launch_bounds__(256) void gemm_bf16_kernel(
        const unsigned short* __restrict__ A,
        const unsigned short* __restrict__ W,
        void* __restrict__ out, int N, int K, int mode, float scale) {
    const int lane = threadIdx.x & 31;
    const int wave = threadIdx.x >> 5;
    const int half = lane >> 4;
    const int lm   = lane & 15;
    const int m_base = blockIdx.y * 64  + (wave >> 2) * 32;
    const int n_base = blockIdx.x * 256 + (wave & 3)  * 64;

    v8f acc[2][4] = {};
    for (int k0 = 0; k0 < K; k0 += 32) {
        Frag a[2], b[4];
#pragma unroll
        for (int i = 0; i < 2; ++i) {
            const unsigned short* p = A + (size_t)(m_base + i*16 + lm) * K + k0 + half*8;
            a[i].u[0] = *(const uint4*)(p);
            a[i].u[1] = *(const uint4*)(p + 16);
        }
#pragma unroll
        for (int j = 0; j < 4; ++j) {
            const unsigned short* p = W + (size_t)(n_base + j*16 + lm) * K + k0 + half*8;
            b[j].u[0] = *(const uint4*)(p);
            b[j].u[1] = *(const uint4*)(p + 16);
        }
        __builtin_prefetch(A + (size_t)(m_base + lm) * K + k0 + 64, 0, 3);
#pragma unroll
        for (int i = 0; i < 2; ++i)
#pragma unroll
            for (int j = 0; j < 4; ++j)
                acc[i][j] = wmma_bf16(a[i], b[j], acc[i][j]);
    }

#pragma unroll
    for (int i = 0; i < 2; ++i) {
#pragma unroll
        for (int j = 0; j < 4; ++j) {
            const int n = n_base + j*16 + lm;
#pragma unroll
            for (int r = 0; r < 8; ++r) {
                const int m = m_base + i*16 + half*8 + r;
                const float val = acc[i][j][r];
                if (mode == 0) {
                    ((float*)out)[(size_t)m * N + n] = val;
                } else {
                    const int b = m >> 11, t = m & (T_ - 1);
                    const int h = n >> 6,  d = n & (DK_ - 1);
                    const unsigned short bv = f2bf(val * scale);
                    if (mode == 3)
                        ((unsigned short*)out)[(((size_t)b*H_ + h)*DK_ + d)*T_ + t] = bv;
                    else
                        ((unsigned short*)out)[(((size_t)b*H_ + h)*T_ + t)*DK_ + d] = bv;
                }
            }
        }
    }
}

// ---------------------------------------------------------------- scores
// attn[bh,t,s] = q_scaled[bh,t,:]·k[bh,s,:] + pq[h,t,:]·pk[h,s,:]
__global__ __launch_bounds__(256) void scores_kernel(
        const unsigned short* __restrict__ q,  const unsigned short* __restrict__ k,
        const unsigned short* __restrict__ pq, const unsigned short* __restrict__ pk,
        float* __restrict__ attn) {
    const int lane = threadIdx.x & 31;
    const int wave = threadIdx.x >> 5;
    const int half = lane >> 4;
    const int lm   = lane & 15;
    const int bh   = blockIdx.z;
    const int h    = bh & (H_ - 1);
    const int t_base = blockIdx.y * 64  + (wave >> 2) * 32;
    const int s_base = blockIdx.x * 128 + (wave & 3)  * 32;
    const unsigned short* qh  = q  + (size_t)bh * T_ * DK_;
    const unsigned short* kh  = k  + (size_t)bh * T_ * DK_;
    const unsigned short* pqh = pq + h * DK_;
    const unsigned short* pkh = pk + h * DK_;

    v8f acc[2][2] = {};
    for (int k0 = 0; k0 < DK_; k0 += 32) {
        Frag qa[2], pa[2], kb[2], pb[2];
#pragma unroll
        for (int i = 0; i < 2; ++i) {
            const unsigned short* p0 = qh + (size_t)(t_base + i*16 + lm) * DK_ + k0 + half*8;
            qa[i].u[0] = *(const uint4*)(p0);
            qa[i].u[1] = *(const uint4*)(p0 + 16);
            const unsigned short* p1 = pqh + (size_t)(t_base + i*16 + lm) * D_ + k0 + half*8;
            pa[i].u[0] = *(const uint4*)(p1);
            pa[i].u[1] = *(const uint4*)(p1 + 16);
        }
#pragma unroll
        for (int j = 0; j < 2; ++j) {
            const unsigned short* p0 = kh + (size_t)(s_base + j*16 + lm) * DK_ + k0 + half*8;
            kb[j].u[0] = *(const uint4*)(p0);
            kb[j].u[1] = *(const uint4*)(p0 + 16);
            const unsigned short* p1 = pkh + (size_t)(s_base + j*16 + lm) * D_ + k0 + half*8;
            pb[j].u[0] = *(const uint4*)(p1);
            pb[j].u[1] = *(const uint4*)(p1 + 16);
        }
#pragma unroll
        for (int i = 0; i < 2; ++i)
#pragma unroll
            for (int j = 0; j < 2; ++j) {
                acc[i][j] = wmma_bf16(qa[i], kb[j], acc[i][j]);
                acc[i][j] = wmma_bf16(pa[i], pb[j], acc[i][j]);
            }
    }

#pragma unroll
    for (int i = 0; i < 2; ++i)
#pragma unroll
        for (int j = 0; j < 2; ++j) {
            const int s = s_base + j*16 + lm;
#pragma unroll
            for (int r = 0; r < 8; ++r) {
                const int t = t_base + i*16 + half*8 + r;
                attn[((size_t)bh * T_ + t) * T_ + s] = acc[i][j][r];
            }
        }
}

// ---------------------------------------------------------------- softmax (rows of length T_)
__global__ __launch_bounds__(256) void softmax_kernel(float* __restrict__ attn) {
    __shared__ float red[256];
    float* p = attn + (size_t)blockIdx.x * T_;
    const int tid = threadIdx.x;

    float vals[T_ / 256];
    float m = -3.0e38f;
#pragma unroll
    for (int i = 0; i < T_ / 256; ++i) {
        vals[i] = p[tid + i * 256];
        m = fmaxf(m, vals[i]);
    }
    red[tid] = m; __syncthreads();
    for (int s = 128; s > 0; s >>= 1) {
        if (tid < s) red[tid] = fmaxf(red[tid], red[tid + s]);
        __syncthreads();
    }
    m = red[0]; __syncthreads();

    float sum = 0.0f;
#pragma unroll
    for (int i = 0; i < T_ / 256; ++i) {
        vals[i] = __expf(vals[i] - m);
        sum += vals[i];
    }
    red[tid] = sum; __syncthreads();
    for (int s = 128; s > 0; s >>= 1) {
        if (tid < s) red[tid] += red[tid + s];
        __syncthreads();
    }
    const float inv = 1.0f / red[0];
#pragma unroll
    for (int i = 0; i < T_ / 256; ++i)
        p[tid + i * 256] = vals[i] * inv;
}

// ---------------------------------------------------------------- AV: out[b,t,h*64+d] = sum_s attn[bh,t,s] * v[bh,s,d]
__global__ __launch_bounds__(256) void av_kernel(
        const float* __restrict__ attn, const unsigned short* __restrict__ vt,
        unsigned short* __restrict__ ao) {
    const int lane = threadIdx.x & 31;
    const int wave = threadIdx.x >> 5;
    const int half = lane >> 4;
    const int lm   = lane & 15;
    const int bh   = blockIdx.z;
    const int b    = bh >> 4, h = bh & 15;
    const int m_base = blockIdx.x * 256 + wave * 32;
    const float*          ah = attn + (size_t)bh * T_ * T_;
    const unsigned short* vh = vt   + (size_t)bh * DK_ * T_;

    v8f acc[2][4] = {};
    for (int k0 = 0; k0 < T_; k0 += 32) {
        Frag a[2], bf[4];
#pragma unroll
        for (int i = 0; i < 2; ++i) {
            const float* p = ah + (size_t)(m_base + i*16 + lm) * T_ + k0 + half*8;
            float4 f0 = *(const float4*)(p);
            float4 f1 = *(const float4*)(p + 4);
            float4 f2 = *(const float4*)(p + 16);
            float4 f3 = *(const float4*)(p + 20);
            float tmp[16] = {f0.x,f0.y,f0.z,f0.w, f1.x,f1.y,f1.z,f1.w,
                             f2.x,f2.y,f2.z,f2.w, f3.x,f3.y,f3.z,f3.w};
#pragma unroll
            for (int e = 0; e < 16; ++e) a[i].h[e] = f2bf(tmp[e]);
        }
#pragma unroll
        for (int j = 0; j < 4; ++j) {
            const unsigned short* p = vh + (size_t)(j*16 + lm) * T_ + k0 + half*8;
            bf[j].u[0] = *(const uint4*)(p);
            bf[j].u[1] = *(const uint4*)(p + 16);
        }
#pragma unroll
        for (int i = 0; i < 2; ++i)
#pragma unroll
            for (int j = 0; j < 4; ++j)
                acc[i][j] = wmma_bf16(a[i], bf[j], acc[i][j]);
    }

#pragma unroll
    for (int i = 0; i < 2; ++i)
#pragma unroll
        for (int j = 0; j < 4; ++j) {
            const int d = j*16 + lm;
#pragma unroll
            for (int r = 0; r < 8; ++r) {
                const int t = m_base + i*16 + half*8 + r;
                ao[((size_t)b * T_ + t) * D_ + h * DK_ + d] = f2bf(acc[i][j][r]);
            }
        }
}

// ---------------------------------------------------------------- host
extern "C" void kernel_launch(void* const* d_in, const int* in_sizes, int n_in,
                              void* d_out, int out_size, void* d_ws, size_t ws_size,
                              hipStream_t stream) {
    (void)in_sizes; (void)n_in; (void)out_size; (void)ws_size;
    const float* x    = (const float*)d_in[0];
    const float* Wq   = (const float*)d_in[1];
    const float* Wk   = (const float*)d_in[2];
    const float* Wv   = (const float*)d_in[3];
    const float* Wout = (const float*)d_in[4];
    const float* peq  = (const float*)d_in[5];
    const float* pek  = (const float*)d_in[6];

    const size_t MK  = (size_t)B_ * T_ * D_;     // 8388608 (x / qkv / attn_out elems)
    const size_t WW  = (size_t)D_ * D_;          // 1048576
    const size_t PP  = (size_t)T_ * D_;          // 2097152 (pos_emb[:T])

    char* ws = (char*)d_ws;
    size_t off = 0;
    unsigned short* xb   = (unsigned short*)(ws + off); off += MK * 2;
    unsigned short* wqb  = (unsigned short*)(ws + off); off += WW * 2;
    unsigned short* wkb  = (unsigned short*)(ws + off); off += WW * 2;
    unsigned short* wvb  = (unsigned short*)(ws + off); off += WW * 2;
    unsigned short* wob  = (unsigned short*)(ws + off); off += WW * 2;
    unsigned short* pqb  = (unsigned short*)(ws + off); off += PP * 2;
    unsigned short* pkb  = (unsigned short*)(ws + off); off += PP * 2;
    unsigned short* qb   = (unsigned short*)(ws + off); off += MK * 2;
    unsigned short* kb   = (unsigned short*)(ws + off); off += MK * 2;
    unsigned short* vtb  = (unsigned short*)(ws + off); off += MK * 2;
    unsigned short* aob  = (unsigned short*)(ws + off); off += MK * 2;

    float* outf = (float*)d_out;
    float* attn = outf + MK;                     // attn tensor region of d_out

    // 1) bf16 conversion passes
    conv_bf16_kernel<<<(int)(MK / 256), 256, 0, stream>>>(x,    xb,  (int)MK);
    conv_bf16_kernel<<<(int)(WW / 256), 256, 0, stream>>>(Wq,   wqb, (int)WW);
    conv_bf16_kernel<<<(int)(WW / 256), 256, 0, stream>>>(Wk,   wkb, (int)WW);
    conv_bf16_kernel<<<(int)(WW / 256), 256, 0, stream>>>(Wv,   wvb, (int)WW);
    conv_bf16_kernel<<<(int)(WW / 256), 256, 0, stream>>>(Wout, wob, (int)WW);
    conv_bf16_kernel<<<(int)(PP / 256), 256, 0, stream>>>(peq,  pqb, (int)PP);
    conv_bf16_kernel<<<(int)(PP / 256), 256, 0, stream>>>(pek,  pkb, (int)PP);

    // 2) QKV projections (q pre-scaled by 1/sqrt(dk))
    dim3 gG(D_ / 256, (B_ * T_) / 64);           // (4, 128)
    gemm_bf16_kernel<<<gG, 256, 0, stream>>>(xb, wqb, qb,  D_, D_, 1, 0.125f);
    gemm_bf16_kernel<<<gG, 256, 0, stream>>>(xb, wkb, kb,  D_, D_, 2, 1.0f);
    gemm_bf16_kernel<<<gG, 256, 0, stream>>>(xb, wvb, vtb, D_, D_, 3, 1.0f);

    // 3) content + positional scores fused into attn region
    dim3 gS(T_ / 128, T_ / 64, B_ * H_);         // (16, 32, 64)
    scores_kernel<<<gS, 256, 0, stream>>>(qb, kb, pqb, pkb, attn);

    // 4) softmax in-place
    softmax_kernel<<<B_ * H_ * T_, 256, 0, stream>>>(attn);

    // 5) attn @ V -> [B,T,D] bf16
    dim3 gAV(T_ / 256, 1, B_ * H_);              // (8, 1, 64)
    av_kernel<<<gAV, 256, 0, stream>>>(attn, vtb, aob);

    // 6) output projection -> first B*T*D floats of d_out
    gemm_bf16_kernel<<<gG, 256, 0, stream>>>(aob, wob, (void*)outf, D_, D_, 0, 1.0f);
}
